// gcn_53601191854663
// MI455X (gfx1250) — matrix operand order repaired and checked by reading, not compiled, not run on hardware
//
#include <hip/hip_runtime.h>
#include <math.h>
#include <stdint.h>

typedef _Float16 v16h __attribute__((ext_vector_type(16)));
typedef float    v8f  __attribute__((ext_vector_type(8)));

#define N_   32
#define T_   256
#define V_   25
#define CIN_ 64
#define FK_  192   // F*K
#define F_   64
#define K_   3

// ---------------- operand-layout index helpers (wave32, 16-bit) ----------------
// A-matrix 16x32 f16 (contiguous k order, matches B): lane = (k/16)*16 + m ; packed pos p = k%16
__device__ __forceinline__ void a_pos(int m, int kidx, int& lane, int& p) {
    int e   = kidx & 1;
    int r4  = (kidx >> 1) & 3;
    int hf  = (kidx >> 3) & 1;
    int rhi = (kidx >> 4) & 1;
    lane = rhi * 16 + m;
    p    = hf * 8 + r4 * 2 + e;
}
// B-matrix 32x16 f16 (contiguous k order, matches A): lanes 0-15 K=0-15, 16-31 K=16-31
__device__ __forceinline__ void b_pos(int kidx, int ncol, int& lane, int& p) {
    lane = ((kidx >> 4) << 4) + ncol;
    p    = kidx & 15;
}

// =====================================================================
// Kernel 1: attention path -> adj[n,k,v,w] = A + B + softmax(C)  (to d_ws)
// one block per n, 256 threads
// =====================================================================
__global__ void attn_adj_kernel(const float* __restrict__ xg,
                                const float* __restrict__ Ag,
                                const float* __restrict__ Bg,
                                const float* __restrict__ Watt,
                                const float* __restrict__ aw,
                                float* __restrict__ adjws) {
    __shared__ float sXb[V_ * CIN_];   // 1600: mean over t
    __shared__ float sG[V_ * 96];      // 2400
    __shared__ float sS1[V_ * K_];     // 75
    __shared__ float sS2[V_ * K_];     // 75
    __shared__ float sE[V_ * V_ * K_]; // 1875

    const int n   = blockIdx.x;
    const int tid = threadIdx.x;

    // xbar = mean_t x[n,t,v,c]
    for (int idx = tid; idx < V_ * CIN_; idx += 256) {
        int v = idx >> 6, c = idx & 63;
        const float* px = xg + (size_t)n * T_ * V_ * CIN_ + v * CIN_ + c;
        float s = 0.f;
        for (int t = 0; t < T_; ++t) s += px[(size_t)t * V_ * CIN_];
        sXb[idx] = s * (1.0f / T_);
    }
    __syncthreads();

    // g = xbar @ W_att  (25x64 @ 64x96)
    for (int idx = tid; idx < V_ * 96; idx += 256) {
        int v = idx / 96, d = idx % 96;
        float s = 0.f;
        for (int c = 0; c < CIN_; ++c) s += sXb[v * CIN_ + c] * Watt[c * 96 + d];
        sG[idx] = s;
    }
    __syncthreads();

    // per (v,k): s1 = <g_t[v,k,:], a_w[:32]> ; s2 = <g_t[v,k,:], a_w[32:]>
    // g_t[v,k,c] = g[v, c*3 + k]
    if (tid < V_ * K_) {
        int v = tid / K_, kk = tid % K_;
        float s1 = 0.f, s2 = 0.f;
        for (int c = 0; c < 32; ++c) {
            float gv = sG[v * 96 + c * K_ + kk];
            s1 += gv * aw[c];
            s2 += gv * aw[32 + c];
        }
        sS1[tid] = s1;
        sS2[tid] = s2;
    }
    __syncthreads();

    // eflat[(i*25+j)*3+k] = lrelu(s1[j,k] + s2[i,k])
    for (int idx = tid; idx < V_ * V_ * K_; idx += 256) {
        int p = idx / K_, kk = idx % K_;
        int i = p / V_, j = p % V_;
        float val = sS1[j * K_ + kk] + sS2[i * K_ + kk];
        sE[idx] = (val > 0.f) ? val : 0.2f * val;
    }
    __syncthreads();

    // raw reshape (n,v*v,k) -> (n,k,v,v); softmax over last dim; adj = A + B + C
    if (tid < K_ * V_) { // 75 rows
        int kk = tid / V_, a = tid % V_;
        int base = kk * (V_ * V_) + a * V_;
        float m = -1e30f;
        for (int b = 0; b < V_; ++b) m = fmaxf(m, sE[base + b]);
        float sum = 0.f;
        for (int b = 0; b < V_; ++b) sum += __expf(sE[base + b] - m);
        float inv = 1.0f / sum;
        for (int b = 0; b < V_; ++b) {
            float C = __expf(sE[base + b] - m) * inv;
            adjws[(size_t)n * (K_ * V_ * V_) + base + b] = Ag[base + b] + Bg[base + b] + C;
        }
    }
}

// =====================================================================
// Kernel 2: fused  h = relu(bn(x@W_conv+b)) ;  y = einsum(h, adj)
// grid (n=32, tchunk=32) ; block 256 = 8 waves ; 8 t's per block, 2 per group
// =====================================================================
__global__ void __launch_bounds__(256)
fused_gcn_kernel(const float* __restrict__ xg,
                 const float* __restrict__ adjws,
                 const float* __restrict__ Wcg,
                 const float* __restrict__ bconv,
                 const float* __restrict__ gamma,
                 const float* __restrict__ beta,
                 const float* __restrict__ mean,
                 const float* __restrict__ var,
                 float* __restrict__ out) {
    // tile = 16 halfs per lane * 32 lanes = 512 halfs = 1024 B
    __shared__ __attribute__((aligned(32))) _Float16 sWc[24 * 512];  // B-layout W_conv (2 ks x 12 nt)
    __shared__ __attribute__((aligned(32))) _Float16 sAdj[6 * 512];  // A-layout AdjT   (2 mt x 3 ks)
    __shared__ __attribute__((aligned(32))) _Float16 sX[8 * 512];    // A-layout x      (4 mt x 2 ks)
    __shared__ __attribute__((aligned(32))) _Float16 sH[24 * 512];   // B-layout Hhat   (2 t  x 3 ks x 4 nt)
    __shared__ float sScale[FK_];
    __shared__ float sShift[FK_];

    const int n     = blockIdx.x;
    const int tbase = blockIdx.y * 8;
    const int tid   = threadIdx.x;
    const int lane  = tid & 31;
    const int wave  = tid >> 5;

    // ---- phase 0: zero pad-sensitive LDS (Hhat K-pad rows, Adj garbage rows) ----
    // Note: sH tiles with ks2==2 (byte ranges [8192,12288) and [20480,24576)) hold the
    // K-padding zeros and are NEVER reused as scratch below.
    for (int i = tid; i < 24 * 512; i += 256) sH[i] = (_Float16)0.f;
    for (int i = tid; i < 6 * 512; i += 256) sAdj[i] = (_Float16)0.f;
    __syncthreads();

    // ---- phase 1: stage constants ----
    // BN fold: scale = gamma*rsqrt(var+eps); shift = beta + (b - mean)*scale
    if (tid < FK_) {
        float s = gamma[tid] * rsqrtf(var[tid] + 1e-3f);
        sScale[tid] = s;
        sShift[tid] = beta[tid] + (bconv[tid] - mean[tid]) * s;
    }
    // W_conv (64 x 192) -> f16 B-layout tiles (ks = c/32, nt = d/16)
    for (int i = tid; i < CIN_ * FK_; i += 256) {
        int c = i / FK_, d = i % FK_;
        int ks = c >> 5, kidx = c & 31;
        int nt = d >> 4, ncol = d & 15;
        int l2, p2;
        b_pos(kidx, ncol, l2, p2);
        sWc[(ks * 12 + nt) * 512 + l2 * 16 + p2] = (_Float16)Wcg[i];
    }
    // adj[n,k,v,w] -> AdjT[w, kv] f16 A-layout tiles (mt = w/16, ks = kv/32)
    for (int i = tid; i < K_ * V_ * V_; i += 256) {
        int kk = i / (V_ * V_);
        int v  = (i / V_) % V_;
        int w  = i % V_;
        int kv = kk * V_ + v;
        int mt = w >> 4, m = w & 15;
        int ks = kv >> 5, kidx = kv & 31;
        int l2, p2;
        a_pos(m, kidx, l2, p2);
        sAdj[(mt * 3 + ks) * 512 + l2 * 16 + p2] =
            (_Float16)adjws[(size_t)n * (K_ * V_ * V_) + i];
    }
    __syncthreads();

    // ---- hoist GEMM2 A-operand (adjacency, block-constant) into registers ----
    const int tl  = wave >> 2;        // t_local 0/1
    const int q   = wave & 3;
    const int mt2 = q & 1;            // M tile (w)
    const int ntb = (q >> 1) * 2;     // two N tiles per wave
    const v16h A0 = *(const v16h*)&sAdj[(mt2 * 3 + 0) * 512 + lane * 16];
    const v16h A1 = *(const v16h*)&sAdj[(mt2 * 3 + 1) * 512 + lane * 16];
    const v16h A2 = *(const v16h*)&sAdj[(mt2 * 3 + 2) * 512 + lane * 16];

    const size_t xbase = (size_t)n * T_ * V_ * CIN_;
    // raw-x scratch overlaps the ks2<2 sH tiles (fully rewritten by GEMM1 each group):
    // bytes [0,8192) and [12288,16896)
    const uint32_t ldsH = (uint32_t)(uintptr_t)(void*)sH;
    char* const sHraw = (char*)sH;

    // ---- group loop: 4 groups of 2 t's ----
    for (int g = 0; g < 4; ++g) {
        const int t0 = tbase + g * 2;
        __syncthreads(); // prev group's sH/sX reads complete

        // ---- async DMA raw x (2*V*CIN f32 = 12800 B) into LDS scratch ----
        const float* xp = xg + xbase + (size_t)t0 * V_ * CIN_;
        const uint64_t gbase = (uint64_t)(uintptr_t)xp;
        for (int i = tid; i < 800; i += 256) {           // 800 x b128
            int b = i * 16;
            uint32_t laddr = ldsH + (uint32_t)(b + ((b < 8192) ? 0 : 4096));
            uint64_t gaddr = gbase + (uint64_t)b;
            asm volatile("global_load_async_to_lds_b128 %0, %1, off"
                         :: "v"(laddr), "v"(gaddr)
                         : "memory");
        }
        // prefetch next group's x slice into cache while DMA runs
        if (g < 3 && tid < 100) {
            __builtin_prefetch(xp + 2 * V_ * CIN_ + tid * 32, 0, 3);
        }
        asm volatile("s_wait_asynccnt 0" ::: "memory");
        __syncthreads(); // all waves' async data landed in LDS

        // ---- convert raw f32 -> f16, scatter into A-layout sX ----
        for (int i = tid; i < 2 * V_ * CIN_; i += 256) {
            int b = i * 4;
            float valf = *(const float*)(sHraw + b + ((b < 8192) ? 0 : 4096));
            int row = i >> 6, c = i & 63;
            int mt = row >> 4, m = row & 15;
            int ks = c >> 5, kidx = c & 31;
            int l2, p2;
            a_pos(m, kidx, l2, p2);
            sX[(mt * 2 + ks) * 512 + l2 * 16 + p2] = (_Float16)valf;
        }
        __syncthreads(); // sX ready; raw scratch reads done (GEMM1 may overwrite)

        // ---- GEMM1: (50x64) @ (64x192), BN+ReLU, scatter to Hhat B-layout ----
        const int half = lane >> 4, ncol = lane & 15;
        for (int id = wave; id < 48; id += 8) {
            int mt = id & 3, nt = id >> 2;
            v16h a0 = *(const v16h*)&sX[(mt * 2 + 0) * 512 + lane * 16];
            v16h a1 = *(const v16h*)&sX[(mt * 2 + 1) * 512 + lane * 16];
            v16h b0 = *(const v16h*)&sWc[(0 * 12 + nt) * 512 + lane * 16];
            v16h b1 = *(const v16h*)&sWc[(1 * 12 + nt) * 512 + lane * 16];
            union { v8f v; float f[8]; } acc;
            acc.v = (v8f){0.f, 0.f, 0.f, 0.f, 0.f, 0.f, 0.f, 0.f};
            acc.v = __builtin_amdgcn_wmma_f32_16x16x32_f16(false, a0, false, b0,
                                                           (short)0, acc.v, false, false);
            acc.v = __builtin_amdgcn_wmma_f32_16x16x32_f16(false, a1, false, b1,
                                                           (short)0, acc.v, false, false);
            int d  = nt * 16 + ncol;
            float sc = sScale[d], sh = sShift[d];
            int kk = d >> 6, f = d & 63;
            int nt2 = f >> 4, nc2 = f & 15;

            auto storeH = [&](int row, float raw) {
                float val = fmaxf(raw * sc + sh, 0.f);
                int tloc = (row >= V_) ? 1 : 0;
                int v  = row - tloc * V_;
                int kv = kk * V_ + v;
                int ks2 = kv >> 5, kidx2 = kv & 31;
                int l2 = ((kidx2 >> 4) << 4) + nc2;
                int p2 = kidx2 & 15;
                sH[(tloc * 12 + ks2 * 4 + nt2) * 512 + l2 * 16 + p2] = (_Float16)val;
            };

            if (mt < 3) {
                // rows mt*16 .. mt*16+15 all valid (< 48 <= 50)
#pragma unroll
                for (int r = 0; r < 8; ++r)
                    storeH(mt * 16 + half * 8 + r, acc.f[r]);
            } else {
                // tile rows 48..63: only rows 48,49 valid (half==0, r<2)
                if (half == 0) {
                    storeH(48, acc.f[0]);
                    storeH(49, acc.f[1]);
                }
            }
        }
        __syncthreads();

        // ---- GEMM2: per t, (25x96-pad) AdjT @ (96x64) Hhat -> y(25x64) ----
        {
            int t = t0 + tl;
            const int nc = lane & 15, hf = lane >> 4;
            float* orow = out + (((size_t)n * T_ + t) * V_) * F_ + nc;
#pragma unroll
            for (int jj = 0; jj < 2; ++jj) {
                int nt2 = ntb + jj;
                v16h b0 = *(const v16h*)&sH[(tl * 12 + 0 * 4 + nt2) * 512 + lane * 16];
                v16h b1 = *(const v16h*)&sH[(tl * 12 + 1 * 4 + nt2) * 512 + lane * 16];
                v16h b2 = *(const v16h*)&sH[(tl * 12 + 2 * 4 + nt2) * 512 + lane * 16];
                union { v8f v; float f[8]; } acc;
                acc.v = (v8f){0.f, 0.f, 0.f, 0.f, 0.f, 0.f, 0.f, 0.f};
                acc.v = __builtin_amdgcn_wmma_f32_16x16x32_f16(false, A0, false, b0,
                                                               (short)0, acc.v, false, false);
                acc.v = __builtin_amdgcn_wmma_f32_16x16x32_f16(false, A1, false, b1,
                                                               (short)0, acc.v, false, false);
                acc.v = __builtin_amdgcn_wmma_f32_16x16x32_f16(false, A2, false, b2,
                                                               (short)0, acc.v, false, false);
                float* ocol = orow + nt2 * 16;
                if (mt2 == 0) {
                    // rows 0..15 all valid
#pragma unroll
                    for (int r = 0; r < 8; ++r)
                        ocol[(size_t)(hf * 8 + r) * F_] = acc.f[r];
                } else {
                    // rows 16..31: half 0 -> 16..23 (valid); half 1 -> 24..31 (only 24)
                    if (hf == 0) {
#pragma unroll
                        for (int r = 0; r < 8; ++r)
                            ocol[(size_t)(16 + r) * F_] = acc.f[r];
                    } else {
                        ocol[(size_t)24 * F_] = acc.f[0];
                    }
                }
            }
        }
    }
}

// =====================================================================
extern "C" void kernel_launch(void* const* d_in, const int* in_sizes, int n_in,
                              void* d_out, int out_size, void* d_ws, size_t ws_size,
                              hipStream_t stream) {
    const float* x     = (const float*)d_in[0];
    const float* Ag    = (const float*)d_in[1];
    const float* Bg    = (const float*)d_in[2];
    const float* Watt  = (const float*)d_in[3];
    const float* aw    = (const float*)d_in[4];
    const float* Wconv = (const float*)d_in[5];
    const float* bconv = (const float*)d_in[6];
    const float* gamma = (const float*)d_in[7];
    const float* beta  = (const float*)d_in[8];
    const float* mean  = (const float*)d_in[9];
    const float* var   = (const float*)d_in[10];
    float* out         = (float*)d_out;
    float* adjws       = (float*)d_ws; // 32*3*25*25 floats = 240 KB

    attn_adj_kernel<<<dim3(N_), 256, 0, stream>>>(x, Ag, Bg, Watt, aw, adjws);
    fused_gcn_kernel<<<dim3(N_, T_ / 8), 256, 0, stream>>>(
        x, adjws, Wconv, bconv, gamma, beta, mean, var, out);
}
